// Action2_17695265259883
// MI455X (gfx1250) — matrix operand order repaired and checked by reading, not compiled, not run on hardware
//
#include <hip/hip_runtime.h>
#include <math.h>

typedef __attribute__((ext_vector_type(16))) _Float16 v16h;
typedef __attribute__((ext_vector_type(8)))  float    v8f;

#define IN_DIM 5
#define HID 64
#define C_MU 0.1f
#define C_NU 0.2f
#define C_SIGMA 0.3f
#define FLOOR_V 1e-4f
#define BN_EPS_V 1e-5f

// ---------------------------------------------------------------------------
// Kernel 1: batch statistics for BatchNorm (mean + rstd per feature column).
// typeVec is (B, IN_DIM); bn and bnc share the same input stats.
// stats layout in ws: [mean[0..4], rstd[0..4]]
// ---------------------------------------------------------------------------
__global__ void bn_stats_kernel(const float* __restrict__ tv,
                                float* __restrict__ stats, int B) {
  __shared__ float ss[256];
  __shared__ float sq[256];
  const int c = blockIdx.x;           // feature column 0..IN_DIM-1
  const int t = threadIdx.x;
  float s = 0.f, q = 0.f;
  for (int r = t; r < B; r += blockDim.x) {
    float v = tv[r * IN_DIM + c];
    s += v;
    q += v * v;
  }
  ss[t] = s;
  sq[t] = q;
  __syncthreads();
  for (int w = 128; w > 0; w >>= 1) {
    if (t < w) { ss[t] += ss[t + w]; sq[t] += sq[t + w]; }
    __syncthreads();
  }
  if (t == 0) {
    float mean = ss[0] / (float)B;
    float var  = sq[0] / (float)B - mean * mean;   // biased, matches reference
    stats[c] = mean;
    stats[IN_DIM + c] = rsqrtf(var + BN_EPS_V);
  }
}

// ---------------------------------------------------------------------------
// Kernel 2: the sequential scan. One wave (32 threads) owns 16 batch rows
// for all N steps. Layer-2 (64x64) of both MLPs runs on WMMA f16->f32.
// ---------------------------------------------------------------------------
__launch_bounds__(32)
__global__ void sde_scan_kernel(
    const float* __restrict__ bm, const float* __restrict__ cn,
    const float* __restrict__ tv, const float* __restrict__ mx,
    const float* __restrict__ mc, const float* __restrict__ initial,
    const float* __restrict__ bng, const float* __restrict__ bnb,
    const float* __restrict__ bncg, const float* __restrict__ bncb,
    const float* __restrict__ w1,  const float* __restrict__ b1,
    const float* __restrict__ w2,  const float* __restrict__ b2,
    const float* __restrict__ w3,  const float* __restrict__ b3,
    const float* __restrict__ wc1, const float* __restrict__ bc1,
    const float* __restrict__ wc2, const float* __restrict__ bc2,
    const float* __restrict__ wc3, const float* __restrict__ bc3,
    const float* __restrict__ stats, float* __restrict__ out,
    int B, int N)
{
  const int l    = threadIdx.x;     // 0..31, wave32
  const int mlo  = l & 15;
  const int hi   = l >> 4;          // 0 or 1 (half of the wave)
  const int b0   = blockIdx.x * 16; // first batch row of this tile
  const float dt = 1.0f / (float)N;

  __shared__ __align__(16) float    sbase_p[16][HID];
  __shared__ __align__(16) float    sbase_c[16][HID];
  __shared__ __align__(16) _Float16 h1p[16][HID];
  __shared__ __align__(16) _Float16 h1c[16][HID];
  __shared__ float w1t[HID], w1x[HID], w1mx[HID], w1mc[HID];
  __shared__ float wc1t[HID], wc1x[HID], wc1mx[HID], wc1mc[HID];
  __shared__ float sbn[16][IN_DIM], sbnc[16][IN_DIM];
  __shared__ float sx[16], sbm[16], sdcn[16], smx[16], smc[16];
  __shared__ float spi[16], sc[16];

  // --- step-varying rows of W1 (t, x, mx, mc = rows 5..8) into LDS
  for (int j = l; j < HID; j += 32) {
    w1t  [j] = w1 [5 * HID + j];  w1x  [j] = w1 [6 * HID + j];
    w1mx [j] = w1 [7 * HID + j];  w1mc [j] = w1 [8 * HID + j];
    wc1t [j] = wc1[5 * HID + j];  wc1x [j] = wc1[6 * HID + j];
    wc1mx[j] = wc1[7 * HID + j];  wc1mc[j] = wc1[8 * HID + j];
  }

  // --- batchnorm outputs for this tile + x0 + j=0 output row
  if (l < 16) {
    int b = b0 + l;
    #pragma unroll
    for (int k = 0; k < IN_DIM; ++k) {
      float v = tv[b * IN_DIM + k];
      float z = (v - stats[k]) * stats[IN_DIM + k];
      sbn [l][k] = z * bng [k] + bnb [k];
      sbnc[l][k] = z * bncg[k] + bncb[k];
    }
    sx[l] = initial[0];
    out[(b * (N + 1)) * 2 + 0] = 0.f;
    out[(b * (N + 1)) * 2 + 1] = initial[0];
  }
  __syncthreads();

  // --- static part of layer-1 preactivation: base = bn_out @ W1[0:5] + b1
  {
    int b = mlo, jb = hi * 32;
    for (int j = 0; j < 32; ++j) {
      int jj = jb + j;
      float a_p = b1[jj], a_c = bc1[jj];
      #pragma unroll
      for (int k = 0; k < IN_DIM; ++k) {
        a_p += sbn [b][k] * w1 [k * HID + jj];
        a_c += sbnc[b][k] * wc1[k * HID + jj];
      }
      sbase_p[b][jj] = a_p;
      sbase_c[b][jj] = a_c;
    }
  }

  // --- W2 as register-resident WMMA B fragments (f32 -> f16).
  // B k order used here (contiguous, not AMD's table order): lane column N = l&15; lanes 0-15 hold K=kb..kb+15,
  // lanes 16-31 hold K=kb+16..kb+31 (2 halves per VGPR => v16h elem e <-> K+e).
  const int koff = hi ? 16 : 0;
  v16h bfp[4][2], bfc[4][2];
  #pragma unroll
  for (int nt = 0; nt < 4; ++nt) {
    #pragma unroll
    for (int kc = 0; kc < 2; ++kc) {
      int n  = nt * 16 + mlo;
      int kb = kc * 32 + koff;
      v16h fp, fc;
      #pragma unroll
      for (int e = 0; e < 16; ++e) {
        fp[e] = (_Float16)w2 [(kb + e) * HID + n];
        fc[e] = (_Float16)wc2[(kb + e) * HID + n];
      }
      bfp[nt][kc] = fp;
      bfc[nt][kc] = fc;
    }
  }

  // --- per-lane layer-2 bias and layer-3 weight (depend only on n = tile+lane)
  float b2r[4], w3r[4], bc2r[4], wc3r[4];
  #pragma unroll
  for (int nt = 0; nt < 4; ++nt) {
    int n = nt * 16 + mlo;
    b2r[nt]  = b2 [n];  w3r[nt]  = w3 [n];
    bc2r[nt] = bc2[n];  wc3r[nt] = wc3[n];
  }
  const float b3s = b3[0], bc3s = bc3[0];
  __syncthreads();

  // ============================ scan over steps ============================
  for (int i = 0; i < N; ++i) {
    const float t = (float)i * dt;

    if (l < 16) {
      int b = b0 + l;
      sbm [l] = bm[b * N + i];
      sdcn[l] = cn[b * (N + 1) + i + 1] - cn[b * (N + 1) + i];
      smx [l] = mx[b * (N + 1) + i];
      smc [l] = mc[b * N + i];
      if (i + 1 < N) {   // pull next step's lines toward the WGP
        __builtin_prefetch(&bm[b * N + i + 1], 0, 3);
        __builtin_prefetch(&cn[b * (N + 1) + i + 2], 0, 3);
        __builtin_prefetch(&mx[b * (N + 1) + i + 1], 0, 3);
        __builtin_prefetch(&mc[b * N + i + 1], 0, 3);
      }
    }
    __syncthreads();

    // layer 1: base + t*w_t + x*w_x + mx*w_mx + mc*w_mc, relu, to f16 LDS
    {
      int b = mlo, jb = hi * 32;
      float xv = sx[b], mxv = smx[b], mcv = smc[b];
      for (int j = 0; j < 32; ++j) {
        int jj = jb + j;
        float hp = sbase_p[b][jj] + t * w1t [jj] + xv * w1x [jj]
                 + mxv * w1mx [jj] + mcv * w1mc [jj];
        float hc = sbase_c[b][jj] + t * wc1t[jj] + xv * wc1x[jj]
                 + mxv * wc1mx[jj] + mcv * wc1mc[jj];
        h1p[b][jj] = (_Float16)fmaxf(hp, 0.f);
        h1c[b][jj] = (_Float16)fmaxf(hc, 0.f);
      }
    }
    __syncthreads();

    // A fragments (contiguous k order used here, not AMD's table order): lane M = l&15; khi = (hi?16:0);
    // v16h elems 0..15 <-> K=khi..khi+15 (matches B fragment order).
    v16h ap[2], ac[2];
    const int khi = hi ? 16 : 0;
    #pragma unroll
    for (int kc = 0; kc < 2; ++kc) {
      int base = kc * 32 + khi;
      v16h fa, fb;
      #pragma unroll
      for (int e = 0; e < 8; ++e) {
        fa[e]     = h1p[mlo][base + e];
        fa[8 + e] = h1p[mlo][base + 8 + e];
        fb[e]     = h1c[mlo][base + e];
        fb[8 + e] = h1c[mlo][base + 8 + e];
      }
      ap[kc] = fa;
      ac[kc] = fb;
    }

    // layer 2 via WMMA, fused bias+relu+w3 partial dot in accumulator regs.
    // C/D layout: reg r, lane l -> m = r + (hi?8:0), n = nt*16 + (l&15).
    float pp[8], pc[8];
    #pragma unroll
    for (int r = 0; r < 8; ++r) { pp[r] = 0.f; pc[r] = 0.f; }
    #pragma unroll
    for (int nt = 0; nt < 4; ++nt) {
      v8f accp = {};
      accp = __builtin_amdgcn_wmma_f32_16x16x32_f16(false, ap[0], false, bfp[nt][0],
                                                    (short)0, accp, false, false);
      accp = __builtin_amdgcn_wmma_f32_16x16x32_f16(false, ap[1], false, bfp[nt][1],
                                                    (short)0, accp, false, false);
      v8f accc = {};
      accc = __builtin_amdgcn_wmma_f32_16x16x32_f16(false, ac[0], false, bfc[nt][0],
                                                    (short)0, accc, false, false);
      accc = __builtin_amdgcn_wmma_f32_16x16x32_f16(false, ac[1], false, bfc[nt][1],
                                                    (short)0, accc, false, false);
      #pragma unroll
      for (int r = 0; r < 8; ++r) {
        pp[r] += fmaxf(accp[r] + b2r[nt], 0.f) * w3r[nt];
        pc[r] += fmaxf(accc[r] + bc2r[nt], 0.f) * wc3r[nt];
      }
    }

    // reduce the n dimension across the 16 lanes of each half
    #pragma unroll
    for (int mask = 8; mask >= 1; mask >>= 1) {
      #pragma unroll
      for (int r = 0; r < 8; ++r) {
        pp[r] += __shfl_xor(pp[r], mask, 32);
        pc[r] += __shfl_xor(pc[r], mask, 32);
      }
    }
    if (mlo == 0) {            // lane 0 -> m=0..7 ; lane 16 -> m=8..15
      int mb = hi * 8;
      #pragma unroll
      for (int r = 0; r < 8; ++r) {
        spi[mb + r] = pp[r] + b3s;
        sc [mb + r] = __expf(pc[r] + bc3s);
      }
    }
    __syncthreads();

    // SDE update + output
    if (l < 16) {
      int b = b0 + l;
      float x  = sx[l];
      float xn = x + spi[l] * x * (C_MU * dt + C_NU * sbm[l] + C_SIGMA * sdcn[l])
                   - sc[l] * x * dt;
      xn = fmaxf(xn - FLOOR_V, 0.f) + FLOOR_V;
      sx[l] = xn;
      out[(b * (N + 1) + i + 1) * 2 + 0] = (float)(i + 1) * dt;
      out[(b * (N + 1) + i + 1) * 2 + 1] = xn;
    }
    __syncthreads();
  }
}

// ---------------------------------------------------------------------------
extern "C" void kernel_launch(void* const* d_in, const int* in_sizes, int n_in,
                              void* d_out, int out_size, void* d_ws, size_t ws_size,
                              hipStream_t stream) {
  const float* bm      = (const float*)d_in[0];
  const float* cn      = (const float*)d_in[1];
  const float* tv      = (const float*)d_in[2];
  const float* mx      = (const float*)d_in[3];
  const float* mc      = (const float*)d_in[4];
  const float* initial = (const float*)d_in[5];
  const float* bng     = (const float*)d_in[6];
  const float* bnb     = (const float*)d_in[7];
  const float* bncg    = (const float*)d_in[8];
  const float* bncb    = (const float*)d_in[9];
  const float* w1  = (const float*)d_in[10];
  const float* b1  = (const float*)d_in[11];
  const float* w2  = (const float*)d_in[12];
  const float* b2  = (const float*)d_in[13];
  const float* w3  = (const float*)d_in[14];
  const float* b3  = (const float*)d_in[15];
  const float* wc1 = (const float*)d_in[16];
  const float* bc1 = (const float*)d_in[17];
  const float* wc2 = (const float*)d_in[18];
  const float* bc2 = (const float*)d_in[19];
  const float* wc3 = (const float*)d_in[20];
  const float* bc3 = (const float*)d_in[21];
  float* out   = (float*)d_out;
  float* stats = (float*)d_ws;        // 10 floats: mean[5], rstd[5]

  const int B = in_sizes[2] / IN_DIM; // 8192
  const int N = in_sizes[0] / B;      // 100

  bn_stats_kernel<<<IN_DIM, 256, 0, stream>>>(tv, stats, B);
  sde_scan_kernel<<<B / 16, 32, 0, stream>>>(
      bm, cn, tv, mx, mc, initial, bng, bnb, bncg, bncb,
      w1, b1, w2, b2, w3, b3, wc1, bc1, wc2, bc2, wc3, bc3,
      stats, out, B, N);
}
